// Attention_13048110645492
// MI455X (gfx1250) — matrix operand order repaired and checked by reading, not compiled, not run on hardware
//
#include <hip/hip_runtime.h>
#include <hip/hip_bf16.h>

#define HIDDEN 512
#define BATCH 32
#define SEQ 4096
#define M_TOTAL (BATCH * SEQ)
#define ROWS_PER_BLOCK 64
#define GRAN_PER_ROW (HIDDEN / 8)          // 64 granules of 8 halfs (16B) per row
#define ROW_BYTES (HIDDEN * 2)             // 1024B per row in LDS

typedef _Float16 half16 __attribute__((ext_vector_type(16)));
typedef _Float16 half8  __attribute__((ext_vector_type(8)));
typedef float    float8 __attribute__((ext_vector_type(8)));
typedef float    f32x4  __attribute__((ext_vector_type(4)));   // native vec for NT loads

union AFrag { half16 v; half8 h[2]; };

// Hardware tanh: CDNA5 TANH transcendental (1 trans-op vs ~20-op libm poly).
__device__ __forceinline__ float fast_tanh(float x) {
#if __has_builtin(__builtin_amdgcn_tanhf)
    return __builtin_amdgcn_tanhf(x);
#elif __has_builtin(__builtin_amdgcn_tanh_f32)
    return __builtin_amdgcn_tanh_f32(x);
#else
    float r;
    asm("v_tanh_f32 %0, %1" : "=v"(r) : "v"(x));
    return r;
#endif
}

// ---------------------------------------------------------------------------
// Prep 1: We (f32, strided inside attn_w) -> Bh (f16, row-major [n][h])
// ---------------------------------------------------------------------------
__global__ void prep_bh(const float* __restrict__ attn_w,
                        _Float16* __restrict__ Bh) {
    const int n = blockIdx.x;
    const float* src = attn_w + (size_t)n * 1024 + 512;
    _Float16* dst = Bh + (size_t)n * HIDDEN;
    for (int h = threadIdx.x; h < HIDDEN; h += blockDim.x)
        dst[h] = (_Float16)src[h];
}

// ---------------------------------------------------------------------------
// Prep 2: HB[b][k] = dot(hidden[b,:], attn_w[k, 0:512]) + attn_b[k]
// ---------------------------------------------------------------------------
__global__ void prep_hb(const float* __restrict__ hidden,
                        const float* __restrict__ attn_w,
                        const float* __restrict__ attn_b,
                        float* __restrict__ HB) {
    __shared__ float hsh[HIDDEN];
    const int b = blockIdx.x;
    for (int i = threadIdx.x; i < HIDDEN; i += blockDim.x)
        hsh[i] = hidden[(size_t)b * HIDDEN + i];
    __syncthreads();
    for (int k = threadIdx.x; k < HIDDEN; k += blockDim.x) {
        const float4* wr = (const float4*)(attn_w + (size_t)k * 1024);
        float acc = attn_b[k];
        #pragma unroll 4
        for (int j = 0; j < HIDDEN / 4; ++j) {
            float4 w4 = wr[j];
            acc += hsh[4*j+0] * w4.x + hsh[4*j+1] * w4.y +
                   hsh[4*j+2] * w4.z + hsh[4*j+3] * w4.w;
        }
        HB[(size_t)b * HIDDEN + k] = acc;
    }
}

// ---------------------------------------------------------------------------
// Main: fused  tanh(enc @ We^T + HB) . v  ->  per-row scores
//
// 64 rows of enc staged to LDS as f16 (64KB) with a 16B-granule XOR swizzle:
//   granule g of row r lives at (g ^ (r & 15)).
// A fragment rows are mt*16 + ln, so the XOR key == ln: the 16 lanes of each
// half-wave land on 16 distinct bank quads -> conflict-free ds_load_b128,
// and the swizzled offset is shared by all 4 M-tiles (mt = 16KB immediate).
//
// K-chunk is the OUTER loop with all 16 accumulator tiles live: each A/B
// fragment is consumed in the iteration that loads it (no hoistable values,
// no spills), every fragment feeds 4 WMMAs, and each block reads the 512KB
// f16 weight table exactly once. 256 v_wmma_f32_16x16x32_f16 per wave.
// ---------------------------------------------------------------------------
__global__ __launch_bounds__(256)
void attn_energy(const float*    __restrict__ enc,
                 const _Float16* __restrict__ Bh,
                 const float*    __restrict__ HB,
                 const float*    __restrict__ vw,
                 float* __restrict__ sc) {
    __shared__ _Float16 Alds[ROWS_PER_BLOCK * HIDDEN];   // exactly 64 KB

    const int tid    = threadIdx.x;
    const int m_base = blockIdx.x * ROWS_PER_BLOCK;
    const int b      = m_base >> 12;                     // / SEQ

    // Stage A block (f32 -> f16, swizzled granules). enc is streamed once:
    // non-temporal loads keep it from evicting the hot weight table.
    for (int i = tid; i < ROWS_PER_BLOCK * GRAN_PER_ROW; i += 256) {
        const int row = i >> 6;
        const int g   = i & 63;
        const f32x4* s4 =
            (const f32x4*)(enc + (size_t)(m_base + row) * HIDDEN + g * 8);
        f32x4 f0 = __builtin_nontemporal_load(s4);
        f32x4 f1 = __builtin_nontemporal_load(s4 + 1);
        half8 hv;
        hv[0] = (_Float16)f0.x; hv[1] = (_Float16)f0.y;
        hv[2] = (_Float16)f0.z; hv[3] = (_Float16)f0.w;
        hv[4] = (_Float16)f1.x; hv[5] = (_Float16)f1.y;
        hv[6] = (_Float16)f1.z; hv[7] = (_Float16)f1.w;
        const int goff = (g ^ (row & 15)) << 4;          // swizzled 16B granule
        *(half8*)((char*)Alds + row * ROW_BYTES + goff) = hv;
    }
    __syncthreads();

    const int lane = tid & 31;
    const int wave = tid >> 5;    // wave w owns N-tiles [w*4, w*4+4)
    const int ln   = lane & 15;
    const int hi   = lane >> 4;   // 0 or 1 (half-wave)

    const float* HBb = HB + (size_t)b * HIDDEN;
    const char*  Ab  = (const char*)Alds + ln * ROW_BYTES;  // row = mt*16 + ln

    float8 acc[4][4] = {};        // [N-tile][M-tile], 16 live 16x16 tiles

    for (int kc = 0; kc < 16; ++kc) {
        // A logical granules this kc (16-bit A 16x32 layout):
        //   lanes 0-15: K=[0..7],[16..23]; lanes 16-31: K=[8..15],[24..31]
        const int g0   = kc * 4 + hi;
        const int off0 = (g0 ^ ln) << 4;                 // shared by all mt
        const int off1 = ((g0 + 2) ^ ln) << 4;

        AFrag a[4];
        #pragma unroll
        for (int mt = 0; mt < 4; ++mt) {
            a[mt].h[0] = *(const half8*)(Ab + mt * 16 * ROW_BYTES + off0);
            a[mt].h[1] = *(const half8*)(Ab + mt * 16 * ROW_BYTES + off1);
        }

        #pragma unroll
        for (int nt = 0; nt < 4; ++nt) {
            const int col = (wave * 4 + nt) * 16 + ln;   // output column (k)
            // One B fragment, reused by 4 WMMAs (4 M-tiles).
            AFrag bu;
            bu.h[0] = *(const half8*)(Bh + (size_t)col * HIDDEN + kc * 32 + hi * 8);
            bu.h[1] = *(const half8*)(Bh + (size_t)col * HIDDEN + kc * 32 + 16 + hi * 8);
            half16 bf = bu.v;
            #pragma unroll
            for (int mt = 0; mt < 4; ++mt)
                acc[nt][mt] = __builtin_amdgcn_wmma_f32_16x16x32_f16(
                    false, a[mt].v, false, bf, (short)0, acc[nt][mt],
                    false, false);
        }
    }

    // Fused epilogue: hardware tanh + v-weighted accumulate.
    // C layout: VGPR i, lanes 0-15 -> (M=i, N=ln); lanes 16-31 -> (M=8+i)
    float part[4][8];             // [M-tile][C-row] v-weighted tanh partials
    #pragma unroll
    for (int mt = 0; mt < 4; ++mt)
        #pragma unroll
        for (int i = 0; i < 8; ++i) part[mt][i] = 0.0f;

    #pragma unroll
    for (int nt = 0; nt < 4; ++nt) {
        const int col = (wave * 4 + nt) * 16 + ln;
        const float hb = HBb[col];
        const float vv = vw[col];
        #pragma unroll
        for (int mt = 0; mt < 4; ++mt)
            #pragma unroll
            for (int i = 0; i < 8; ++i)
                part[mt][i] += vv * fast_tanh(acc[nt][mt][i] + hb);
    }

    // All waves done reading Alds -> safe to alias it for the score reduction.
    __syncthreads();
    float* ssc = (float*)Alds;    // ssc[wave][64 local rows], 2KB

    #pragma unroll
    for (int mt = 0; mt < 4; ++mt) {
        #pragma unroll
        for (int i = 0; i < 8; ++i) {
            float r = part[mt][i];
            r += __shfl_xor(r, 1, 32);   // masks < 16: reduction stays inside
            r += __shfl_xor(r, 2, 32);   // each 16-lane half (wave32)
            r += __shfl_xor(r, 4, 32);
            r += __shfl_xor(r, 8, 32);
            if (ln == 0)
                ssc[wave * ROWS_PER_BLOCK + mt * 16 + hi * 8 + i] = r;
        }
    }
    __syncthreads();

    // Combine the 8 per-wave partials (deterministic fixed order).
    if (tid < ROWS_PER_BLOCK) {
        float s = 0.0f;
        #pragma unroll
        for (int w = 0; w < 8; ++w) s += ssc[w * ROWS_PER_BLOCK + tid];
        sc[m_base + tid] = s;
    }
}

// ---------------------------------------------------------------------------
// Softmax over S per batch.
// ---------------------------------------------------------------------------
__global__ void softmax_k(const float* __restrict__ sc,
                          float* __restrict__ out) {
    __shared__ float red[256];
    const int b   = blockIdx.x;
    const int tid = threadIdx.x;
    const float* s = sc + (size_t)b * SEQ;
    float* o = out + (size_t)b * SEQ;

    float m = -1e30f;
    for (int i = tid; i < SEQ; i += 256) m = fmaxf(m, s[i]);
    red[tid] = m; __syncthreads();
    for (int st = 128; st > 0; st >>= 1) {
        if (tid < st) red[tid] = fmaxf(red[tid], red[tid + st]);
        __syncthreads();
    }
    m = red[0]; __syncthreads();

    float sum = 0.0f;
    for (int i = tid; i < SEQ; i += 256) {
        float e = expf(s[i] - m);
        o[i] = e;
        sum += e;
    }
    red[tid] = sum; __syncthreads();
    for (int st = 128; st > 0; st >>= 1) {
        if (tid < st) red[tid] += red[tid + st];
        __syncthreads();
    }
    const float inv = 1.0f / red[0];
    for (int i = tid; i < SEQ; i += 256) o[i] *= inv;
}

// ---------------------------------------------------------------------------
extern "C" void kernel_launch(void* const* d_in, const int* in_sizes, int n_in,
                              void* d_out, int out_size, void* d_ws, size_t ws_size,
                              hipStream_t stream) {
    const float* hidden = (const float*)d_in[0];   // (32, 512)
    const float* enc    = (const float*)d_in[1];   // (32, 4096, 512)
    const float* attn_w = (const float*)d_in[2];   // (512, 1024)
    const float* attn_b = (const float*)d_in[3];   // (512,)
    const float* v_w    = (const float*)d_in[4];   // (1, 512)

    char* ws = (char*)d_ws;
    _Float16* Bh = (_Float16*)ws;                          // 512 KB
    float*    HB = (float*)(ws + 512 * 1024);              //  64 KB
    float*    sc = (float*)(ws + 512 * 1024 + 64 * 1024);  // 512 KB

    prep_bh<<<HIDDEN, 256, 0, stream>>>(attn_w, Bh);
    prep_hb<<<BATCH, 256, 0, stream>>>(hidden, attn_w, attn_b, HB);
    attn_energy<<<M_TOTAL / ROWS_PER_BLOCK, 256, 0, stream>>>(enc, Bh, HB, v_w, sc);
    softmax_k<<<BATCH, 256, 0, stream>>>(sc, (float*)d_out);
}
